// UnifiedSparseUNet_87823491269141
// MI455X (gfx1250) — matrix operand order repaired and checked by reading, not compiled, not run on hardware
//
#include <hip/hip_runtime.h>
#include <hip/hip_bf16.h>
#include <cstdint>

// ---------------------------------------------------------------------------
// Sparse 3D U-Net + heads + tiny transformer for MI455X (gfx1250, wave32).
// Convs are implicit GEMMs on v_wmma_f32_16x16x32_f16 (f16 in, f32 accum).
// A-tiles are staged into LDS with GLOBAL_LOAD_ASYNC_TO_LDS_B128 (ASYNCcnt),
// D-tiles are written back through LDS as coalesced b128 stores.
// ---------------------------------------------------------------------------

typedef __attribute__((ext_vector_type(16))) _Float16 v16h;
typedef __attribute__((ext_vector_type(8)))  float    v8f;

#define NB   8        // batch
#define NACT 4096     // active voxels per sample

// ---- CDNA5 async global->LDS copy helpers (CDNA5 ISA) ---
__device__ __forceinline__ void async_ld16(uint32_t lds_off, const void* gptr) {
  asm volatile("global_load_async_to_lds_b128 %0, %1, off"
               :: "v"(lds_off), "v"((uint64_t)(uintptr_t)gptr)
               : "memory");
}
__device__ __forceinline__ void async_wait0() {
  asm volatile("s_wait_asynccnt 0x0" ::: "memory");
}

// ---------------------------------------------------------------------------
// utility kernels
// ---------------------------------------------------------------------------
__global__ void k_fill_i32(int* __restrict__ p, int v, int n) {
  int i = blockIdx.x * blockDim.x + threadIdx.x;
  if (i < n) p[i] = v;
}

// scatter active voxels -> dense index map (level1), occupancy flags lvl2/3
__global__ void k_scatter(const float* __restrict__ features,
                          const int* __restrict__ coords,
                          _Float16* __restrict__ feat0,
                          int* __restrict__ idxmap1,
                          int* __restrict__ list1,
                          int* __restrict__ occ2,
                          int* __restrict__ occ3) {
  int i = blockIdx.x * blockDim.x + threadIdx.x;
  if (i >= NB * NACT) return;
  int b  = i >> 12;
  int cx = coords[i * 3 + 0], cy = coords[i * 3 + 1], cz = coords[i * 3 + 2];
  int site = (cx * 64 + cy) * 64 + cz;
  idxmap1[b * 262144 + site] = i & 4095;
  list1[i] = site;
  feat0[i] = (_Float16)features[i];
  int s2 = ((cx >> 1) * 32 + (cy >> 1)) * 32 + (cz >> 1);
  atomicOr(&occ2[b * 32768 + s2], 1);
  int s3 = ((cx >> 2) * 16 + (cy >> 2)) * 16 + (cz >> 2);
  atomicOr(&occ3[b * 4096 + s3], 1);
}

// compact occupancy flags into per-sample active lists + index maps
__global__ void k_compact(const int* __restrict__ occ,
                          int* __restrict__ idxmap,
                          int* __restrict__ list,
                          int* __restrict__ count,
                          int sites_per_b) {
  int j = blockIdx.x * blockDim.x + threadIdx.x;
  if (j >= NB * sites_per_b) return;
  int b = j / sites_per_b;
  int s = j - b * sites_per_b;
  if (occ[j]) {
    int idx = atomicAdd(&count[b], 1);
    idxmap[j] = idx;
    list[b * NACT + idx] = s;
  }
}

// ---------------------------------------------------------------------------
// weight pre-swizzle: pack W[K][COUT] (f32) into the per-lane WMMA B-fragment
// layout, converted to f16.  bpack[((kc*NT+nt)*32+lane)*16+e] =
//     W[kc*32 + e + 16*(lane>>4)][nt*16 + (lane&15)]
// subm layout: W[k][o] = w[(o*CIN + k%CIN)*27 + k/CIN]      (w is O,I,3,3,3)
// down/up    : W[k][o] = w[k*COUT + o]                      (w is 2,2,2,I,O)
// ---------------------------------------------------------------------------
__global__ void k_prep_w(const float* __restrict__ w, _Float16* __restrict__ bpack,
                         int CIN, int COUT, int KPAD, int TAPS, int subm) {
  int i = blockIdx.x * blockDim.x + threadIdx.x;
  int NT = COUT >> 4;
  int total = (KPAD >> 5) * NT * 512;
  if (i >= total) return;
  int e    = i & 15;
  int lane = (i >> 4) & 31;
  int nt   = (i >> 9) % NT;
  int kc   = i / (512 * NT);
  int k = kc * 32 + e + ((lane >> 4) << 4);
  int n = nt * 16 + (lane & 15);
  float val = 0.f;
  int KT = TAPS * CIN;
  if (k < KT) {
    int tap = k / CIN, ci = k - tap * CIN;
    val = subm ? w[(n * CIN + ci) * 27 + tap]
               : w[(tap * CIN + ci) * COUT + n];
  }
  bpack[i] = (_Float16)val;
}

// ---------------------------------------------------------------------------
// generic sparse conv as implicit GEMM on WMMA.
//  MODE 0: submanifold 3x3x3 (same level, 27 taps)
//  MODE 1: down 2x2x2 s2 (input dim = 2*ODIM, 8 taps = children)
//  MODE 2: inverse up 2x2x2 (input dim = ODIM/2, tap selected by site parity)
// ---------------------------------------------------------------------------
template<int CIN, int COUT, int TAPS, int MODE, int ODIM, int KPAD>
__global__ __launch_bounds__(32 * (COUT / 16))
void k_conv(const _Float16* __restrict__ xin,
            _Float16* __restrict__ yout,
            const _Float16* __restrict__ bpack,
            const int* __restrict__ list,
            const int* __restrict__ counts,
            const int* __restrict__ idxmap_in,
            const _Float16* __restrict__ zrow) {
  constexpr int NT   = COUT / 16;
  constexpr int BD   = 32 * NT;
  constexpr int IDIM = (MODE == 0) ? ODIM : (MODE == 1 ? ODIM * 2 : ODIM / 2);
  constexpr int INS  = IDIM * IDIM * IDIM;
  constexpr int KT   = TAPS * CIN;

  __shared__ _Float16 As[16][KPAD];
  __shared__ int sh_site[16];
  __shared__ int sh_nb[TAPS][16];

  const int b = blockIdx.y, tile = blockIdx.x, tid = threadIdx.x;
  const int n = counts ? counts[b] : NACT;
  if (tile * 16 >= n) return;

  if (tid < 16) {
    int row = tile * 16 + tid;
    sh_site[tid] = (row < n) ? list[b * NACT + row] : -1;
  }
  if constexpr (KT < KPAD) {               // zero-pad K tail once
    constexpr int PADC = KPAD - KT;
    for (int z = tid; z < 16 * PADC; z += BD)
      As[z / PADC][KT + (z % PADC)] = (_Float16)0;
  }
  __syncthreads();

  // -------- decode ALL neighbor indices up front (one barrier) -------------
  for (int t = tid; t < TAPS * 16; t += BD) {
    int tap = t >> 4, m = t & 15;
    int s = sh_site[m];
    int idx = -1;
    if (s >= 0) {
      int d = s / (ODIM * ODIM), h = (s / ODIM) % ODIM, w = s % ODIM;
      if constexpr (MODE == 0) {
        int nd = d + tap / 9 - 1, nh = h + (tap / 3) % 3 - 1, nw = w + tap % 3 - 1;
        if ((unsigned)nd < (unsigned)ODIM && (unsigned)nh < (unsigned)ODIM &&
            (unsigned)nw < (unsigned)ODIM)
          idx = idxmap_in[b * INS + (nd * IDIM + nh) * IDIM + nw];
      } else if constexpr (MODE == 1) {
        int nd = 2 * d + ((tap >> 2) & 1), nh = 2 * h + ((tap >> 1) & 1),
            nw = 2 * w + (tap & 1);
        idx = idxmap_in[b * INS + (nd * IDIM + nh) * IDIM + nw];
      } else {
        int mytap = ((d & 1) << 2) | ((h & 1) << 1) | (w & 1);
        if (tap == mytap)
          idx = idxmap_in[b * INS + ((d >> 1) * IDIM + (h >> 1)) * IDIM + (w >> 1)];
      }
    }
    sh_nb[tap][m] = idx;
  }
  __syncthreads();

  // -------- stage A tile (16 x TAPS*CIN) into LDS --------------------------
  if constexpr (CIN >= 16) {
    // async global->LDS, 16B per lane-chunk, invalid rows pull from zero row
    constexpr int CPR = CIN / 8;           // 16B chunks per (row,tap)
    for (int c = tid; c < TAPS * 16 * CPR; c += BD) {
      int tap = c / (16 * CPR);
      int rem = c - tap * 16 * CPR;
      int m = rem / CPR, sub = rem - m * CPR;
      int idx = sh_nb[tap][m];
      const _Float16* src = (idx >= 0)
          ? xin + ((size_t)(b * NACT + idx)) * CIN + sub * 8
          : zrow + sub * 8;
      async_ld16((uint32_t)(uintptr_t)&As[m][tap * CIN + sub * 8], src);
    }
    async_wait0();                         // each wave drains its own ASYNCcnt
  } else {
    for (int e = tid; e < 16 * KT; e += BD) {
      int m = e / KT, k = e - m * KT;
      int tap = k / CIN, ci = k - tap * CIN;
      int idx = sh_nb[tap][m];
      As[m][k] = (idx >= 0) ? xin[((size_t)(b * NACT + idx)) * CIN + ci]
                            : (_Float16)0;
    }
  }
  __syncthreads();

  // -------- WMMA sweep over K ----------------------------------------------
  const int lane = tid & 31, nt = tid >> 5;
  const int half = lane >> 4, mrow = lane & 15;
  v8f acc = {};
  for (int kc = 0; kc < KPAD / 32; ++kc) {
    v16h a, bb;
    const _Float16* arow = &As[mrow][kc * 32];
    // A layout (matches bpack): element e holds K = e + 16*half
#pragma unroll
    for (int e = 0; e < 16; ++e) a[e] = arow[16 * half + e];
    const _Float16* bp = bpack + ((size_t)(kc * NT + nt) * 32 + lane) * 16;
    __builtin_prefetch(bp + (size_t)NT * 512, 0, 0);   // next K-chunk weights
#pragma unroll
    for (int e = 0; e < 16; ++e) bb[e] = bp[e];
    acc = __builtin_amdgcn_wmma_f32_16x16x32_f16(false, a, false, bb,
                                                 (short)0, acc, false, false);
  }
  __syncthreads();                         // all waves done reading As

  // -------- stage D in LDS, write back as coalesced b128 -------------------
  _Float16* Ds = &As[0][0];                // 16*COUT <= 16*KPAD always
#pragma unroll
  for (int r = 0; r < 8; ++r)
    Ds[(8 * half + r) * COUT + nt * 16 + mrow] = (_Float16)acc[r];
  __syncthreads();
  constexpr int OPR = COUT / 8;            // 16B chunks per output row
  for (int c = tid; c < 16 * OPR; c += BD) {
    int m = c / OPR, sub = c - m * OPR;
    *(float4*)(yout + ((size_t)(b * NACT + tile * 16 + m)) * COUT + sub * 8) =
        *(const float4*)(Ds + m * COUT + sub * 8);
  }
}

// ---------------------------------------------------------------------------
// per-voxel heads: f_org -> probs, f_dir -> tokens, f_rec -> output
// ---------------------------------------------------------------------------
__global__ void k_heads(const _Float16* __restrict__ x,
                        const float* __restrict__ w_org, const float* __restrict__ b_org,
                        const float* __restrict__ w_dir, const float* __restrict__ b_dir,
                        const float* __restrict__ w_rec, const float* __restrict__ b_rec,
                        const float* __restrict__ w_pool, const float* __restrict__ b_pool,
                        float* __restrict__ out_frec,
                        float* __restrict__ probs,
                        float* __restrict__ toks) {
  int i = blockIdx.x * blockDim.x + threadIdx.x;
  if (i >= NB * NACT) return;
  float xg[16];
#pragma unroll
  for (int c = 0; c < 16; ++c) xg[c] = (float)x[(size_t)i * 16 + c];
  float forg = b_org[0], frec = b_rec[0];
  float fd[3] = {b_dir[0], b_dir[1], b_dir[2]};
#pragma unroll
  for (int c = 0; c < 16; ++c) {
    forg += xg[c] * w_org[c];
    frec += xg[c] * w_rec[c];
    fd[0] += xg[c] * w_dir[c * 3 + 0];
    fd[1] += xg[c] * w_dir[c * 3 + 1];
    fd[2] += xg[c] * w_dir[c * 3 + 2];
  }
  out_frec[i] = frec;
  float pv = 1.f / (1.f + expf(-forg));
  pv = fminf(fmaxf(pv, 1e-6f), 1.0f);
  probs[i] = pv;
  int b = i >> 12, t = i & 4095;
  if (t < 500) {
    float* tk = toks + ((size_t)b * 500 + t) * 16;
#pragma unroll
    for (int j = 0; j < 16; ++j)
      tk[j] = b_pool[j] + fd[0] * w_pool[j] + fd[1] * w_pool[16 + j] +
              fd[2] * w_pool[32 + j];
  }
}

// probability-weighted centroid (deterministic fixed-order tree reduction)
__global__ void k_centroid(const float* __restrict__ probs,
                           const int* __restrict__ coords,
                           float* __restrict__ out) {
  __shared__ float red[256][4];
  int b = blockIdx.x, t = threadIdx.x;
  float sp = 0, sx = 0, sy = 0, sz = 0;
  for (int i = t; i < NACT; i += 256) {
    float p = probs[b * NACT + i];
    const int* c = coords + ((size_t)b * NACT + i) * 3;
    sp += p; sx += p * (float)c[0]; sy += p * (float)c[1]; sz += p * (float)c[2];
  }
  red[t][0] = sp; red[t][1] = sx; red[t][2] = sy; red[t][3] = sz;
  __syncthreads();
  for (int s = 128; s > 0; s >>= 1) {
    if (t < s)
      for (int k = 0; k < 4; ++k) red[t][k] += red[t + s][k];
    __syncthreads();
  }
  if (t == 0) {
    float d = red[0][0] + 1e-8f;
    out[b * 3 + 0] = red[0][1] / d - 32.0f;
    out[b * 3 + 1] = red[0][2] / d - 32.0f;
    out[b * 3 + 2] = red[0][3] / d - 32.0f;
  }
}

// ---------------------------------------------------------------------------
// tiny transformer (d_model=16, 4 heads, dh=4, T=500, post-norm)
// ---------------------------------------------------------------------------
__global__ void k_qkv(const float* __restrict__ toks, const float* __restrict__ w,
                      const float* __restrict__ bias, float* __restrict__ qkv) {
  int i = blockIdx.x * blockDim.x + threadIdx.x;
  if (i >= NB * 500) return;
  float x[16];
#pragma unroll
  for (int c = 0; c < 16; ++c) x[c] = toks[(size_t)i * 16 + c];
  for (int j = 0; j < 48; ++j) {
    float s = bias[j];
#pragma unroll
    for (int c = 0; c < 16; ++c) s += x[c] * w[c * 48 + j];
    qkv[(size_t)i * 48 + j] = s;
  }
}

__global__ void k_attn(const float* __restrict__ qkv, float* __restrict__ attn) {
  int bh = blockIdx.x;
  int b = bh >> 2, h = bh & 3;
  int t = threadIdx.x;
  if (t >= 500) return;
  const float* base = qkv + (size_t)b * 500 * 48;
  float q[4];
#pragma unroll
  for (int d = 0; d < 4; ++d) q[d] = base[t * 48 + h * 4 + d];
  float m = -1e30f, l = 0.f, acc[4] = {0, 0, 0, 0};
  for (int j = 0; j < 500; ++j) {
    const float* kj = base + j * 48 + 16 + h * 4;
    float s = 0.5f * (q[0] * kj[0] + q[1] * kj[1] + q[2] * kj[2] + q[3] * kj[3]);
    float mn = fmaxf(m, s);
    float sc = expf(m - mn);
    float p  = expf(s - mn);
    l = l * sc + p;
    const float* vj = base + j * 48 + 32 + h * 4;
#pragma unroll
    for (int d = 0; d < 4; ++d) acc[d] = acc[d] * sc + p * vj[d];
    m = mn;
  }
  float inv = 1.f / l;
#pragma unroll
  for (int d = 0; d < 4; ++d)
    attn[((size_t)b * 500 + t) * 16 + h * 4 + d] = acc[d] * inv;
}

__device__ inline void ln16(const float* in, const float* g, const float* bb,
                            float* out) {
  float mu = 0;
#pragma unroll
  for (int c = 0; c < 16; ++c) mu += in[c];
  mu *= (1.f / 16.f);
  float var = 0;
#pragma unroll
  for (int c = 0; c < 16; ++c) { float d = in[c] - mu; var += d * d; }
  var *= (1.f / 16.f);
  float r = rsqrtf(var + 1e-5f);
#pragma unroll
  for (int c = 0; c < 16; ++c) out[c] = (in[c] - mu) * r * g[c] + bb[c];
}

__global__ void k_post(const float* __restrict__ toks, const float* __restrict__ attn,
                       const float* __restrict__ w_o, const float* __restrict__ b_o,
                       const float* __restrict__ ln1g, const float* __restrict__ ln1b,
                       const float* __restrict__ w_ff1, const float* __restrict__ b_ff1,
                       const float* __restrict__ w_ff2, const float* __restrict__ b_ff2,
                       const float* __restrict__ ln2g, const float* __restrict__ ln2b,
                       float* __restrict__ xrows) {
  int i = blockIdx.x * blockDim.x + threadIdx.x;
  if (i >= NB * 500) return;
  float x0[16], a[16], t1[16], y[16], z[16], o[16];
#pragma unroll
  for (int c = 0; c < 16; ++c) {
    x0[c] = toks[(size_t)i * 16 + c];
    a[c]  = attn[(size_t)i * 16 + c];
  }
  for (int n = 0; n < 16; ++n) {
    float s = b_o[n];
#pragma unroll
    for (int c = 0; c < 16; ++c) s += a[c] * w_o[c * 16 + n];
    t1[n] = x0[n] + s;
  }
  ln16(t1, ln1g, ln1b, y);
#pragma unroll
  for (int n = 0; n < 16; ++n) z[n] = b_ff2[n];
  for (int j = 0; j < 32; ++j) {
    float u = b_ff1[j];
#pragma unroll
    for (int c = 0; c < 16; ++c) u += y[c] * w_ff1[c * 32 + j];
    u = fmaxf(u, 0.f);
#pragma unroll
    for (int n = 0; n < 16; ++n) z[n] += u * w_ff2[j * 16 + n];
  }
#pragma unroll
  for (int n = 0; n < 16; ++n) z[n] += y[n];
  ln16(z, ln2g, ln2b, o);
#pragma unroll
  for (int c = 0; c < 16; ++c) xrows[(size_t)i * 16 + c] = o[c];
}

__global__ void k_final(const float* __restrict__ xrows,
                        const float* __restrict__ w_fin, const float* __restrict__ b_fin,
                        float* __restrict__ out) {
  int b = blockIdx.x, t = threadIdx.x;
  __shared__ float gf[16];
  if (t < 16) {
    float s = 0;
    for (int j = 0; j < 500; ++j) s += xrows[((size_t)b * 500 + j) * 16 + t];
    gf[t] = s * (1.f / 500.f);
  }
  __syncthreads();
  if (t == 0) {
    float v[3];
    for (int k = 0; k < 3; ++k) {
      float s = b_fin[k];
      for (int c = 0; c < 16; ++c) s += gf[c] * w_fin[c * 3 + k];
      v[k] = s;
    }
    float nrm = sqrtf(v[0] * v[0] + v[1] * v[1] + v[2] * v[2]);
    if (nrm >= 1e-8f && nrm == nrm) {
      float inv = 1.f / fmaxf(nrm, 1e-8f);
      out[24 + b * 3 + 0] = v[0] * inv;
      out[24 + b * 3 + 1] = v[1] * inv;
      out[24 + b * 3 + 2] = v[2] * inv;
    } else {
      out[24 + b * 3 + 0] = 1.f;
      out[24 + b * 3 + 1] = 0.f;
      out[24 + b * 3 + 2] = 0.f;
    }
  }
}

// ---------------------------------------------------------------------------
// host launcher
// ---------------------------------------------------------------------------
extern "C" void kernel_launch(void* const* d_in, const int* in_sizes, int n_in,
                              void* d_out, int out_size, void* d_ws, size_t ws_size,
                              hipStream_t stream) {
  const float* features = (const float*)d_in[0];
  const int*   coords   = (const int*)d_in[1];
  const float* w_enc1   = (const float*)d_in[2];
  const float* w_down1  = (const float*)d_in[3];
  const float* w_enc2   = (const float*)d_in[4];
  const float* w_down2  = (const float*)d_in[5];
  const float* w_bot    = (const float*)d_in[6];
  const float* w_up2    = (const float*)d_in[7];
  const float* w_dec2   = (const float*)d_in[8];
  const float* w_up1    = (const float*)d_in[9];
  const float* w_dec1   = (const float*)d_in[10];
  const float* w_org    = (const float*)d_in[11];
  const float* b_org    = (const float*)d_in[12];
  const float* w_dir    = (const float*)d_in[13];
  const float* b_dir    = (const float*)d_in[14];
  const float* w_rec    = (const float*)d_in[15];
  const float* b_rec    = (const float*)d_in[16];
  const float* w_pool   = (const float*)d_in[17];
  const float* b_pool   = (const float*)d_in[18];
  const float* w_qkv    = (const float*)d_in[19];
  const float* b_qkv    = (const float*)d_in[20];
  const float* w_o      = (const float*)d_in[21];
  const float* b_o      = (const float*)d_in[22];
  const float* ln1_g    = (const float*)d_in[23];
  const float* ln1_b    = (const float*)d_in[24];
  const float* w_ff1    = (const float*)d_in[25];
  const float* b_ff1    = (const float*)d_in[26];
  const float* w_ff2    = (const float*)d_in[27];
  const float* b_ff2    = (const float*)d_in[28];
  const float* ln2_g    = (const float*)d_in[29];
  const float* ln2_b    = (const float*)d_in[30];
  const float* w_fin    = (const float*)d_in[31];
  const float* b_fin    = (const float*)d_in[32];
  float* out = (float*)d_out;

  // ---- workspace carve-out ----
  char* base = (char*)d_ws;
  size_t off = 0;
  auto A = [&](size_t bytes) -> char* {
    char* p = base + off;
    off += (bytes + 255) & ~(size_t)255;
    return p;
  };
  _Float16* feat0 = (_Float16*)A((size_t)NB * NACT * 2);
  _Float16* f1a   = (_Float16*)A((size_t)NB * NACT * 16 * 2);
  _Float16* f1b   = (_Float16*)A((size_t)NB * NACT * 16 * 2);
  _Float16* f2a   = (_Float16*)A((size_t)NB * NACT * 32 * 2);
  _Float16* f2b   = (_Float16*)A((size_t)NB * NACT * 32 * 2);
  _Float16* f3a   = (_Float16*)A((size_t)NB * NACT * 64 * 2);
  _Float16* f3b   = (_Float16*)A((size_t)NB * NACT * 64 * 2);
  int* idxmap1 = (int*)A((size_t)NB * 262144 * 4);
  int* idxmap2 = (int*)A((size_t)NB * 32768 * 4);
  int* idxmap3 = (int*)A((size_t)NB * 4096 * 4);
  int* list1   = (int*)A((size_t)NB * NACT * 4);
  int* list2   = (int*)A((size_t)NB * NACT * 4);
  int* list3   = (int*)A((size_t)NB * NACT * 4);
  int* occ2    = (int*)A((size_t)NB * 32768 * 4);
  int* occ3    = (int*)A((size_t)NB * 4096 * 4);
  int* counts  = (int*)A(64);   // [0..7]=n2, [8..15]=n3
  _Float16* zrow = (_Float16*)A(128);   // 64 zero halves for invalid gathers
  _Float16* bp_enc1  = (_Float16*)A((size_t)32   * 1 * 32);
  _Float16* bp_down1 = (_Float16*)A((size_t)128  * 2 * 32);
  _Float16* bp_enc2  = (_Float16*)A((size_t)864  * 2 * 32);
  _Float16* bp_down2 = (_Float16*)A((size_t)256  * 4 * 32);
  _Float16* bp_bot   = (_Float16*)A((size_t)1728 * 4 * 32);
  _Float16* bp_up2   = (_Float16*)A((size_t)512  * 2 * 32);
  _Float16* bp_dec2  = (_Float16*)A((size_t)864  * 2 * 32);
  _Float16* bp_up1   = (_Float16*)A((size_t)256  * 1 * 32);
  _Float16* bp_dec1  = (_Float16*)A((size_t)448  * 1 * 32);
  float* probs = (float*)A((size_t)NB * NACT * 4);
  float* toks  = (float*)A((size_t)NB * 500 * 16 * 4);
  float* qkvb  = (float*)A((size_t)NB * 500 * 48 * 4);
  float* attnb = (float*)A((size_t)NB * 500 * 16 * 4);
  float* xrows = (float*)A((size_t)NB * 500 * 16 * 4);
  if (off > ws_size) return;  // workspace too small; nothing safe to do

  auto G = [](int n, int bd) { return (n + bd - 1) / bd; };

  // ---- init maps / flags / counters / zero row ----
  k_fill_i32<<<G(NB * 262144, 256), 256, 0, stream>>>(idxmap1, -1, NB * 262144);
  k_fill_i32<<<G(NB * 32768, 256), 256, 0, stream>>>(idxmap2, -1, NB * 32768);
  k_fill_i32<<<G(NB * 4096, 256), 256, 0, stream>>>(idxmap3, -1, NB * 4096);
  k_fill_i32<<<G(NB * 32768, 256), 256, 0, stream>>>(occ2, 0, NB * 32768);
  k_fill_i32<<<G(NB * 4096, 256), 256, 0, stream>>>(occ3, 0, NB * 4096);
  k_fill_i32<<<1, 64, 0, stream>>>(counts, 0, 16);
  k_fill_i32<<<1, 64, 0, stream>>>((int*)zrow, 0, 32);

  // ---- build sparse structures ----
  k_scatter<<<G(NB * NACT, 256), 256, 0, stream>>>(features, coords, feat0,
                                                   idxmap1, list1, occ2, occ3);
  k_compact<<<G(NB * 32768, 256), 256, 0, stream>>>(occ2, idxmap2, list2, counts, 32768);
  k_compact<<<G(NB * 4096, 256), 256, 0, stream>>>(occ3, idxmap3, list3, counts + 8, 4096);

  // ---- pre-swizzle weights into WMMA B-fragment layout ----
  auto prep = [&](const float* w, _Float16* bp, int CIN, int COUT, int KPAD,
                  int TAPS, int subm) {
    int total = (KPAD >> 5) * (COUT >> 4) * 512;
    k_prep_w<<<G(total, 256), 256, 0, stream>>>(w, bp, CIN, COUT, KPAD, TAPS, subm);
  };
  prep(w_enc1,  bp_enc1,  1,  16, 32,   27, 1);
  prep(w_down1, bp_down1, 16, 32, 128,  8,  0);
  prep(w_enc2,  bp_enc2,  32, 32, 864,  27, 1);
  prep(w_down2, bp_down2, 32, 64, 256,  8,  0);
  prep(w_bot,   bp_bot,   64, 64, 1728, 27, 1);
  prep(w_up2,   bp_up2,   64, 32, 512,  8,  0);
  prep(w_dec2,  bp_dec2,  32, 32, 864,  27, 1);
  prep(w_up1,   bp_up1,   32, 16, 256,  8,  0);
  prep(w_dec1,  bp_dec1,  16, 16, 448,  27, 1);

  // ---- conv chain (WMMA implicit GEMM) ----
  dim3 cg(256, NB);
  k_conv<1, 16, 27, 0, 64, 32><<<cg, 32, 0, stream>>>(feat0, f1a, bp_enc1, list1, nullptr, idxmap1, zrow);
  k_conv<16, 32, 8, 1, 32, 128><<<cg, 64, 0, stream>>>(f1a, f2a, bp_down1, list2, counts, idxmap1, zrow);
  k_conv<32, 32, 27, 0, 32, 864><<<cg, 64, 0, stream>>>(f2a, f2b, bp_enc2, list2, counts, idxmap2, zrow);
  k_conv<32, 64, 8, 1, 16, 256><<<cg, 128, 0, stream>>>(f2b, f3a, bp_down2, list3, counts + 8, idxmap2, zrow);
  k_conv<64, 64, 27, 0, 16, 1728><<<cg, 128, 0, stream>>>(f3a, f3b, bp_bot, list3, counts + 8, idxmap3, zrow);
  k_conv<64, 32, 8, 2, 32, 512><<<cg, 64, 0, stream>>>(f3b, f2a, bp_up2, list2, counts, idxmap3, zrow);
  k_conv<32, 32, 27, 0, 32, 864><<<cg, 64, 0, stream>>>(f2a, f2b, bp_dec2, list2, counts, idxmap2, zrow);
  k_conv<32, 16, 8, 2, 64, 256><<<cg, 32, 0, stream>>>(f2b, f1a, bp_up1, list1, nullptr, idxmap2, zrow);
  k_conv<16, 16, 27, 0, 64, 448><<<cg, 32, 0, stream>>>(f1a, f1b, bp_dec1, list1, nullptr, idxmap1, zrow);

  // ---- heads, centroid, transformer ----
  k_heads<<<G(NB * NACT, 256), 256, 0, stream>>>(f1b, w_org, b_org, w_dir, b_dir,
                                                 w_rec, b_rec, w_pool, b_pool,
                                                 out + 48, probs, toks);
  k_centroid<<<NB, 256, 0, stream>>>(probs, coords, out);
  k_qkv<<<G(NB * 500, 128), 128, 0, stream>>>(toks, w_qkv, b_qkv, qkvb);
  k_attn<<<NB * 4, 512, 0, stream>>>(qkvb, attnb);
  k_post<<<G(NB * 500, 128), 128, 0, stream>>>(toks, attnb, w_o, b_o, ln1_g, ln1_b,
                                               w_ff1, b_ff1, w_ff2, b_ff2,
                                               ln2_g, ln2_b, xrows);
  k_final<<<NB, 32, 0, stream>>>(xrows, w_fin, b_fin, out);
}
